// TEmb_77283641524426
// MI455X (gfx1250) — matrix operand order repaired and checked by reading, not compiled, not run on hardware
//
#include <hip/hip_runtime.h>

typedef __attribute__((ext_vector_type(16))) _Float16 v16h;
typedef __attribute__((ext_vector_type(8)))  float    v8f;
typedef __attribute__((ext_vector_type(2)))  _Float16 h2;

#define DIM     512
#define OUT_CH  1024
#define M_TILE  128
#define THREADS 256

union Frag16 { v16h v; uint4 u[2]; };

// ---------------- W: f32 -> f16 (1 MB, stays hot in 192MB L2) ----------------
__global__ void w_convert_kernel(const float* __restrict__ W,
                                 _Float16* __restrict__ Wh, int n) {
    int i = blockIdx.x * blockDim.x + threadIdx.x;
    if (i < n) Wh[i] = (_Float16)W[i];
}

// ---------------- emb(LDS, f16) -> WMMA GEMM -> bias+SiLU -> f32 out ---------
__global__ __launch_bounds__(THREADS)
void temb_gemm_kernel(const float* __restrict__ T,
                      const _Float16* __restrict__ Wh,
                      const float* __restrict__ bias,
                      float* __restrict__ out) {
    extern __shared__ _Float16 ldsA[];          // [M_TILE][DIM] f16 = 128 KB

    const int tid     = threadIdx.x;
    const int wg_row0 = blockIdx.x * M_TILE;

    // ---- Phase 1: build the 128x512 embedding tile once, in f16, in LDS ----
    {
        const float cfreq = -0.03597789059f;    // -(2/512)*ln(10000)
        h2* lds2 = (h2*)ldsA;                   // [M_TILE][DIM/2]
        #pragma unroll 4
        for (int it = 0; it < (M_TILE * (DIM / 2)) / THREADS; ++it) {
            int idx = tid + it * THREADS;
            int row = idx >> 8;                 // / (DIM/2)
            int p   = idx & (DIM / 2 - 1);
            float t  = T[wg_row0 + row];
            float f0 = __expf((float)(2 * p)     * cfreq);
            float f1 = __expf((float)(2 * p + 1) * cfreq);
            h2 val;
            val.x = (_Float16)__sinf(t * f0);   // even column -> sin
            val.y = (_Float16)__cosf(t * f1);   // odd  column -> cos
            lds2[row * (DIM / 2) + p] = val;
        }
    }
    __syncthreads();

    // ---- Phase 2: each wave: 16 rows x 64 cols per N-tile, 16 N-tiles ------
    const int wave  = tid >> 5;
    const int lane  = tid & 31;
    const int lhalf = lane & 15;
    const int ksel  = lane >> 4;
    const int rowA  = wave * 16 + lhalf;        // A row within the M-tile
    const int koffA = ksel * 8;                 // A: lanes16-31 take K+8 / K+24
    const int koffB = ksel * 8;                 // B: lanes16-31 take K+8 / K+24

    for (int nt = 0; nt < OUT_CH / 64; ++nt) {
        const int n_base = nt * 64;
        v8f acc[4];
        #pragma unroll
        for (int j = 0; j < 4; ++j) acc[j] = (v8f){0.f,0.f,0.f,0.f,0.f,0.f,0.f,0.f};

        for (int k0 = 0; k0 < DIM; k0 += 32) {
            // A fragment (16x32 f16) from LDS: two ds_load_b128 per lane
            Frag16 a;
            const _Float16* ap = ldsA + rowA * DIM + k0 + koffA;
            a.u[0] = *(const uint4*)(ap);        // K = k0+koffA .. +7
            a.u[1] = *(const uint4*)(ap + 16);   // K = k0+16+koffA .. +7

            #pragma unroll
            for (int j = 0; j < 4; ++j) {
                // B fragment (32x16 f16): column n = row n of W (B = W^T)
                Frag16 bf;
                const _Float16* bp =
                    Wh + (size_t)(n_base + j * 16 + lhalf) * DIM + k0 + koffB;
                bf.u[0] = *(const uint4*)(bp);       // K = k0+koffB .. +7
                bf.u[1] = *(const uint4*)(bp + 16);  // K = k0+16+koffB .. +7
                acc[j] = __builtin_amdgcn_wmma_f32_16x16x32_f16(
                    false, a.v, false, bf.v, (short)0, acc[j], false, false);
            }
        }

        // ---- epilogue: bias + SiLU + f32 store (C/D layout per ISA) --------
        #pragma unroll
        for (int j = 0; j < 4; ++j) {
            const int col = n_base + j * 16 + lhalf;
            const float bv = bias[col];
            #pragma unroll
            for (int r = 0; r < 8; ++r) {
                const int row = wg_row0 + wave * 16 + r + ksel * 8;
                float y = acc[j][r] + bv;
                y = y / (1.0f + __expf(-y));    // SiLU
                out[(size_t)row * OUT_CH + col] = y;
            }
        }
    }
}

extern "C" void kernel_launch(void* const* d_in, const int* in_sizes, int n_in,
                              void* d_out, int out_size, void* d_ws, size_t ws_size,
                              hipStream_t stream) {
    const float* T = (const float*)d_in[0];     // [B]
    const float* W = (const float*)d_in[1];     // [OUT_CH, DIM]
    const float* b = (const float*)d_in[2];     // [OUT_CH]
    float* out = (float*)d_out;                 // [B, OUT_CH]
    _Float16* Wh = (_Float16*)d_ws;             // 1 MB f16 copy of W

    const int nW = OUT_CH * DIM;
    w_convert_kernel<<<(nW + 255) / 256, 256, 0, stream>>>(W, Wh, nW);

    const int Brows = in_sizes[0];              // 65536
    const size_t lds_bytes = (size_t)M_TILE * DIM * sizeof(_Float16); // 128 KB
    temb_gemm_kernel<<<dim3(Brows / M_TILE), THREADS, lds_bytes, stream>>>(
        T, Wh, b, out);
}
